// NeuroBranch_65446711656867
// MI455X (gfx1250) — matrix operand order repaired and checked by reading, not compiled, not run on hardware
//
#include <hip/hip_runtime.h>
#include <math.h>

// ---------------------------------------------------------------------------
// NeuroSAT-style GNN for MI455X (gfx1250, wave32, WMMA).
// GEMMs via v_wmma_f32_16x16x32_f16 (f16 in, f32 accumulate), 32-row blocks
// with B-fragment register reuse (1 weight fragment -> 2 WMMAs).
// Message passing: float4 row gathers + f32 L2 atomics.
// ---------------------------------------------------------------------------

typedef __attribute__((ext_vector_type(16))) _Float16 v16h;
typedef __attribute__((ext_vector_type(8)))  float    v8f;

#define D128 128
#define NVARS 16000
#define NCLAUSES 64000
#define NLITS 32000
#define NROUNDS 4

// ---------------------------------------------------------------------------
// Fragment loaders.
// A (16x32 f16) per CDNA5 ISA: lanes 0-15 -> M=lane, K base 0; lanes 16-31 ->
// M=lane-16, K base 8. Halves h<8: K=base+h, h>=8: K=base+8+h.
// Two 16B LDS chunks per lane -> 2x ds_load_b128.
// ---------------------------------------------------------------------------
__device__ inline v16h ldsA(const _Float16* __restrict__ xs, int stride_h,
                            int kb, int lane) {
    const int m    = lane & 15;
    const int base = kb * 32 + ((lane & 16) ? 8 : 0);
    const unsigned int* p = (const unsigned int*)(xs + (size_t)m * stride_h);
    const int b2 = base >> 1;
    union { v16h v; unsigned int u[8]; } r;
    r.u[0] = p[b2 + 0];  r.u[1] = p[b2 + 1];
    r.u[2] = p[b2 + 2];  r.u[3] = p[b2 + 3];
    r.u[4] = p[b2 + 8];  r.u[5] = p[b2 + 9];
    r.u[6] = p[b2 + 10]; r.u[7] = p[b2 + 11];
    return r.v;
}

// B (32x16 f16): lane L -> N = L%16; halves h: K = (L<16 ? 0 : 8) + (h<8 ? h : h+8).
// Pre-packed so one fragment is 32 contiguous bytes per lane (2x b128).
__device__ inline v16h ldB(const _Float16* __restrict__ packed, int frag,
                           int lane) {
    union { v16h v; uint4 u[2]; } r;
    const uint4* p = (const uint4*)(packed + ((size_t)frag * 32 + lane) * 16);
    r.u[0] = p[0];
    r.u[1] = p[1];
    return r.v;
}

// ---------------------------------------------------------------------------
// Utility kernels
// ---------------------------------------------------------------------------
__global__ void zero_f32(float4* __restrict__ p, int n4) {
    const float4 z = {0.f, 0.f, 0.f, 0.f};
    for (int i = blockIdx.x * blockDim.x + threadIdx.x; i < n4;
         i += gridDim.x * blockDim.x)
        p[i] = z;
}

__global__ void init_state(float4* __restrict__ p, int n4,
                           const float* __restrict__ scale) {
    const float s = *scale;
    const float4 v = {s, s, s, s};
    for (int i = blockIdx.x * blockDim.x + threadIdx.x; i < n4;
         i += gridDim.x * blockDim.x)
        p[i] = v;
}

// Pack row-major f32 W[K][N] into B-fragment f16 order:
// out[((t*KB+kb)*32+lane)*16+h] = W[kb*32 + (lane<16?0:8) + (h<8?h:h+8)][t*16 + lane%16]
__global__ void pack_b(const float* __restrict__ W, _Float16* __restrict__ out,
                       int K, int N) {
    const int total = K * N;
    const int KB = K >> 5;
    for (int idx = blockIdx.x * blockDim.x + threadIdx.x; idx < total;
         idx += gridDim.x * blockDim.x) {
        const int h    = idx & 15;
        const int lane = (idx >> 4) & 31;
        const int rem  = idx >> 9;
        const int kb   = rem % KB;
        const int t    = rem / KB;
        const int n = t * 16 + (lane & 15);
        const int k = kb * 32 + ((lane & 16) ? 8 : 0) + (h < 8 ? h : h + 8);
        out[idx] = (_Float16)W[(size_t)k * N + n];
    }
}

// ---------------------------------------------------------------------------
// Edge scatter: dst[sidx[e]] += src[gidx[e]] * scale  (128-wide rows).
// One wave per edge: float4 gather (global_load_b128), 4 f32 atomics/lane.
// ---------------------------------------------------------------------------
__global__ __launch_bounds__(256) void scatter_msgs(
    const float* __restrict__ src, const int* __restrict__ gidx,
    const int* __restrict__ sidx, float* __restrict__ dst,
    const float* __restrict__ scale_p, int nnz) {
    const int e = blockIdx.x * 8 + (threadIdx.x >> 5);
    if (e >= nnz) return;
    const int lane = threadIdx.x & 31;
    const float s = *scale_p;
    const int g = gidx[e];
    const int d = sidx[e];
    const float4 v = ((const float4*)(src + (size_t)g * D128))[lane];
    float* drow = dst + (size_t)d * D128 + lane * 4;
    atomicAdd(drow + 0, v.x * s);
    atomicAdd(drow + 1, v.y * s);
    atomicAdd(drow + 2, v.z * s);
    atomicAdd(drow + 3, v.w * s);
}

// ---------------------------------------------------------------------------
// Two-layer MLP + normalize(ddof=1) + residual for a 32-row block.
// X = [state | msgs (| flip)] (K0 = 256 or 384). 8 waves; wave w owns output
// columns [16w,16w+16) for BOTH 16-row tiles -> each B fragment feeds 2 WMMAs.
// ---------------------------------------------------------------------------
template <int K0>
__global__ __launch_bounds__(256) void mlp_round(
    const float* __restrict__ state, const float* __restrict__ msgs,
    const float* __restrict__ flipsrc, int half_rows,
    const _Float16* __restrict__ W0p, const float* __restrict__ b0,
    const _Float16* __restrict__ W1p, const float* __restrict__ b1,
    float* __restrict__ out_state) {
    constexpr int KB0 = K0 / 32;
    constexpr int XS = K0 + 8;           // LDS row stride (halves, mult of 8)
    __shared__ _Float16 xs[32 * XS];     // staged f16 input tile (32 rows)
    __shared__ _Float16 h1[32 * 136];    // relu(layer0) f16
    __shared__ float    h2[32 * 132];    // layer1 output f32

    const int tid = threadIdx.x;
    const int lane = tid & 31;
    const int w = tid >> 5;
    const int row0 = blockIdx.x * 32;

    // ---- stage input tile: float4 global loads -> packed f16 b64 LDS stores
    for (int i = tid; i < 32 * (K0 / 4); i += 256) {
        const int m = i / (K0 / 4);
        const int k = (i - m * (K0 / 4)) * 4;    // multiple of 4
        const int row = row0 + m;
        float4 v;
        if (k < 128) {
            v = *(const float4*)(state + (size_t)row * D128 + k);
        } else if (k < 256) {
            v = *(const float4*)(msgs + (size_t)row * D128 + (k - 128));
        } else {
            if constexpr (K0 > 256) {
                const int fr = (row < half_rows) ? row + half_rows
                                                 : row - half_rows;
                v = *(const float4*)(flipsrc + (size_t)fr * D128 + (k - 256));
            } else {
                v = make_float4(0.f, 0.f, 0.f, 0.f);
            }
        }
        union { _Float16 h[4]; uint2 u; } t;
        t.h[0] = (_Float16)v.x; t.h[1] = (_Float16)v.y;
        t.h[2] = (_Float16)v.z; t.h[3] = (_Float16)v.w;
        *(uint2*)(&xs[m * XS + k]) = t.u;        // (m*XS+k)*2 is 8B aligned
    }
    __syncthreads();

    // ---- layer 0: [32 x K0] @ [K0 x 128], ReLU; B fragment reused 2x ----
    v8f acc0 = {}, acc1 = {};
    for (int kb = 0; kb < KB0; ++kb) {
        v16h b  = ldB(W0p, w * KB0 + kb, lane);
        v16h a0 = ldsA(xs, XS, kb, lane);
        v16h a1 = ldsA(xs + 16 * XS, XS, kb, lane);
        acc0 = __builtin_amdgcn_wmma_f32_16x16x32_f16(
            false, a0, false, b, (short)0, acc0, false, false);
        acc1 = __builtin_amdgcn_wmma_f32_16x16x32_f16(
            false, a1, false, b, (short)0, acc1, false, false);
    }
    {
        const float bias = b0[w * 16 + (lane & 15)];
        const int mbase = (lane & 16) ? 8 : 0;
        const int n = w * 16 + (lane & 15);
#pragma unroll
        for (int i = 0; i < 8; ++i) {
            float v0 = acc0[i] + bias;
            float v1 = acc1[i] + bias;
            h1[(mbase + i) * 136 + n]      = (_Float16)(v0 > 0.f ? v0 : 0.f);
            h1[(mbase + i + 16) * 136 + n] = (_Float16)(v1 > 0.f ? v1 : 0.f);
        }
    }
    __syncthreads();

    // ---- layer 1: [32 x 128] @ [128 x 128]; B fragment reused 2x ----
    acc0 = (v8f){}; acc1 = (v8f){};
    for (int kb = 0; kb < 4; ++kb) {
        v16h b  = ldB(W1p, w * 4 + kb, lane);
        v16h a0 = ldsA(h1, 136, kb, lane);
        v16h a1 = ldsA(h1 + 16 * 136, 136, kb, lane);
        acc0 = __builtin_amdgcn_wmma_f32_16x16x32_f16(
            false, a0, false, b, (short)0, acc0, false, false);
        acc1 = __builtin_amdgcn_wmma_f32_16x16x32_f16(
            false, a1, false, b, (short)0, acc1, false, false);
    }
    {
        const float bias = b1[w * 16 + (lane & 15)];
        const int mbase = (lane & 16) ? 8 : 0;
        const int n = w * 16 + (lane & 15);
#pragma unroll
        for (int i = 0; i < 8; ++i) {
            h2[(mbase + i) * 132 + n]      = acc0[i] + bias;
            h2[(mbase + i + 16) * 132 + n] = acc1[i] + bias;
        }
    }
    __syncthreads();

    // ---- normalize (ddof=1) + residual; wave w handles rows w+8rr ----
    for (int rr = 0; rr < 4; ++rr) {
        const int m = w + rr * 8;
        const float4 hv = *(const float4*)(&h2[m * 132 + lane * 4]);
        float s  = hv.x + hv.y + hv.z + hv.w;
        float ss = hv.x * hv.x + hv.y * hv.y + hv.z * hv.z + hv.w * hv.w;
#pragma unroll
        for (int off = 16; off > 0; off >>= 1) {
            s  += __shfl_xor(s, off, 32);
            ss += __shfl_xor(ss, off, 32);
        }
        const float mean = s * (1.f / 128.f);
        float var = (ss - 128.f * mean * mean) * (1.f / 127.f);
        const float inv = rsqrtf(var > 0.f ? var : 1e-30f);
        const size_t row = (size_t)(row0 + m);
        const float4 rv = *(const float4*)(state + row * D128 + lane * 4);
        float4 o;
        o.x = (hv.x - mean) * inv + rv.x;
        o.y = (hv.y - mean) * inv + rv.y;
        o.z = (hv.z - mean) * inv + rv.z;
        o.w = (hv.w - mean) * inv + rv.w;
        *(float4*)(out_state + row * D128 + lane * 4) = o;
    }
}

// ---------------------------------------------------------------------------
// Final scoring: V = [L[v] | L[v+NVARS]] -> relu(V@Vw0+Vb0) . Vw1 + Vb1
// 32 vars per block, B fragment reused across both row tiles.
// ---------------------------------------------------------------------------
__global__ __launch_bounds__(256) void score_kernel(
    const float* __restrict__ L, const _Float16* __restrict__ Vw0p,
    const float* __restrict__ Vb0, const float* __restrict__ Vw1,
    const float* __restrict__ Vb1, float* __restrict__ out) {
    __shared__ _Float16 xs[32 * 264];
    __shared__ float h1f[32 * 132];
    const int tid = threadIdx.x;
    const int lane = tid & 31;
    const int w = tid >> 5;
    const int row0 = blockIdx.x * 32;

    for (int i = tid; i < 32 * 64; i += 256) {
        const int m = i >> 6;
        const int k = (i & 63) * 4;
        const int var = row0 + m;
        float4 v = (k < 128)
                       ? *(const float4*)(L + (size_t)var * D128 + k)
                       : *(const float4*)(L + (size_t)(var + NVARS) * D128 +
                                          (k - 128));
        union { _Float16 h[4]; uint2 u; } t;
        t.h[0] = (_Float16)v.x; t.h[1] = (_Float16)v.y;
        t.h[2] = (_Float16)v.z; t.h[3] = (_Float16)v.w;
        *(uint2*)(&xs[m * 264 + k]) = t.u;
    }
    __syncthreads();

    v8f acc0 = {}, acc1 = {};
    for (int kb = 0; kb < 8; ++kb) {
        v16h b  = ldB(Vw0p, w * 8 + kb, lane);
        v16h a0 = ldsA(xs, 264, kb, lane);
        v16h a1 = ldsA(xs + 16 * 264, 264, kb, lane);
        acc0 = __builtin_amdgcn_wmma_f32_16x16x32_f16(
            false, a0, false, b, (short)0, acc0, false, false);
        acc1 = __builtin_amdgcn_wmma_f32_16x16x32_f16(
            false, a1, false, b, (short)0, acc1, false, false);
    }
    {
        const float bias = Vb0[w * 16 + (lane & 15)];
        const int mbase = (lane & 16) ? 8 : 0;
        const int n = w * 16 + (lane & 15);
#pragma unroll
        for (int i = 0; i < 8; ++i) {
            float v0 = acc0[i] + bias;
            float v1 = acc1[i] + bias;
            h1f[(mbase + i) * 132 + n]      = v0 > 0.f ? v0 : 0.f;
            h1f[(mbase + i + 16) * 132 + n] = v1 > 0.f ? v1 : 0.f;
        }
    }
    __syncthreads();

    for (int rr = 0; rr < 4; ++rr) {
        const int m = w + rr * 8;
        const float4 hv = *(const float4*)(&h1f[m * 132 + lane * 4]);
        const float4 wv = *(const float4*)(Vw1 + lane * 4);
        float s = hv.x * wv.x + hv.y * wv.y + hv.z * wv.z + hv.w * wv.w;
#pragma unroll
        for (int off = 16; off > 0; off >>= 1) s += __shfl_xor(s, off, 32);
        if (lane == 0) out[row0 + m] = s + Vb1[0];
    }
}

// ---------------------------------------------------------------------------
// Host launcher
// ---------------------------------------------------------------------------
extern "C" void kernel_launch(void* const* d_in, const int* in_sizes, int n_in,
                              void* d_out, int out_size, void* d_ws,
                              size_t ws_size, hipStream_t stream) {
    (void)n_in; (void)out_size; (void)ws_size;

    const int nnz = in_sizes[2] / 2;
    const int* cidx = (const int*)d_in[2];
    const int* lidx = cidx + nnz;
    const float* Cw0 = (const float*)d_in[3];
    const float* Cb0 = (const float*)d_in[4];
    const float* Cw1 = (const float*)d_in[5];
    const float* Cb1 = (const float*)d_in[6];
    const float* Lw0 = (const float*)d_in[7];
    const float* Lb0 = (const float*)d_in[8];
    const float* Lw1 = (const float*)d_in[9];
    const float* Lb1 = (const float*)d_in[10];
    const float* Vw0 = (const float*)d_in[11];
    const float* Vb0 = (const float*)d_in[12];
    const float* Vw1 = (const float*)d_in[13];
    const float* Vb1 = (const float*)d_in[14];
    const float* Lis = (const float*)d_in[15];
    const float* Cis = (const float*)d_in[16];
    const float* LCs = (const float*)d_in[17];
    const float* CLs = (const float*)d_in[18];

    // -------- workspace carving (256B aligned regions) --------
    char* ws = (char*)d_ws;
    auto carve = [&](size_t bytes) -> void* {
        void* p = (void*)ws;
        ws += (bytes + 255) & ~(size_t)255;
        return p;
    };
    float* L0 = (float*)carve((size_t)NLITS * D128 * 4);
    float* L1 = (float*)carve((size_t)NLITS * D128 * 4);
    float* C  = (float*)carve((size_t)NCLAUSES * D128 * 4);
    float* LC = (float*)carve((size_t)NCLAUSES * D128 * 4);
    float* CL = (float*)carve((size_t)NLITS * D128 * 4);
    _Float16* Cw0p = (_Float16*)carve((size_t)NROUNDS * 256 * 128 * 2);
    _Float16* Cw1p = (_Float16*)carve((size_t)NROUNDS * 128 * 128 * 2);
    _Float16* Lw0p = (_Float16*)carve((size_t)NROUNDS * 384 * 128 * 2);
    _Float16* Lw1p = (_Float16*)carve((size_t)NROUNDS * 128 * 128 * 2);
    _Float16* Vw0p = (_Float16*)carve((size_t)256 * 128 * 2);

    // -------- pack weights into B-fragment f16 layout --------
    for (int r = 0; r < NROUNDS; ++r) {
        pack_b<<<128, 256, 0, stream>>>(Cw0 + (size_t)r * 256 * 128,
                                        Cw0p + (size_t)r * 256 * 128, 256, 128);
        pack_b<<<64, 256, 0, stream>>>(Cw1 + (size_t)r * 128 * 128,
                                       Cw1p + (size_t)r * 128 * 128, 128, 128);
        pack_b<<<192, 256, 0, stream>>>(Lw0 + (size_t)r * 384 * 128,
                                        Lw0p + (size_t)r * 384 * 128, 384, 128);
        pack_b<<<64, 256, 0, stream>>>(Lw1 + (size_t)r * 128 * 128,
                                       Lw1p + (size_t)r * 128 * 128, 128, 128);
    }
    pack_b<<<128, 256, 0, stream>>>(Vw0, Vw0p, 256, 128);

    // -------- init states --------
    init_state<<<2048, 256, 0, stream>>>((float4*)L0, NLITS * D128 / 4, Lis);
    init_state<<<2048, 256, 0, stream>>>((float4*)C, NCLAUSES * D128 / 4, Cis);

    float* Lcur = L0;
    float* Lnext = L1;
    const int scatter_blocks = (nnz + 7) / 8;

    for (int r = 0; r < NROUNDS; ++r) {
        // LC_msgs = segsum(L[lidx] -> cidx) * LC_scale
        zero_f32<<<2048, 256, 0, stream>>>((float4*)LC, NCLAUSES * D128 / 4);
        scatter_msgs<<<scatter_blocks, 256, 0, stream>>>(Lcur, lidx, cidx, LC,
                                                         LCs, nnz);
        // C = normalize(MLP([C | LC])) + C   (in-place, row-local)
        mlp_round<256><<<NCLAUSES / 32, 256, 0, stream>>>(
            C, LC, nullptr, 0, Cw0p + (size_t)r * 256 * 128,
            Cb0 + (size_t)r * 128, Cw1p + (size_t)r * 128 * 128,
            Cb1 + (size_t)r * 128, C);
        // CL_msgs = segsum(C[cidx] -> lidx) * CL_scale
        zero_f32<<<2048, 256, 0, stream>>>((float4*)CL, NLITS * D128 / 4);
        scatter_msgs<<<scatter_blocks, 256, 0, stream>>>(C, cidx, lidx, CL,
                                                         CLs, nnz);
        // Lnext = normalize(MLP([L | CL | Lflip])) + L  (ping-pong: flip reads)
        mlp_round<384><<<NLITS / 32, 256, 0, stream>>>(
            Lcur, CL, Lcur, NVARS, Lw0p + (size_t)r * 384 * 128,
            Lb0 + (size_t)r * 128, Lw1p + (size_t)r * 128 * 128,
            Lb1 + (size_t)r * 128, Lnext);
        float* t = Lcur; Lcur = Lnext; Lnext = t;
    }

    score_kernel<<<NVARS / 32, 256, 0, stream>>>(Lcur, Vw0p, Vb0, Vw1, Vb1,
                                                 (float*)d_out);
}
